// SpatialSelfAttentionLayer_25254407700927
// MI455X (gfx1250) — compile-verified, matrix operand order checked by reading, not run on hardware
//
#include <hip/hip_runtime.h>

#define B_   16
#define L_   512
#define D_   768
#define H_   12
#define DH_  64
#define SPD_ 5

typedef __attribute__((ext_vector_type(16))) __bf16        v16bf;
typedef __attribute__((ext_vector_type(8)))  float         v8f;
typedef __attribute__((ext_vector_type(4)))  unsigned int  u32x4;

union FragU { u32x4 u[2]; v16bf v; };

// CDNA5 16-bit A/B WMMA fragment: lane half lh = lane>>4 picks k base 8*lh,
// elements k = kb..kb+7 and kb+16..kb+23 (two 16B chunks). p = row base at k0.
__device__ __forceinline__ v16bf load_frag(const __bf16* p, int laneHalf) {
    FragU f;
    const char* base = (const char*)p + laneHalf * 16;
    f.u[0] = *(const u32x4*)(base);
    f.u[1] = *(const u32x4*)(base + 32);
    return f.v;
}

__device__ __forceinline__ v8f wmma_bf16(v16bf a, v16bf b, v8f c) {
    return __builtin_amdgcn_wmma_f32_16x16x32_bf16(
        false, a, false, b, (short)0, c, false, false);
}

// Async global -> LDS 16B copy (GLOBAL_LOAD_ASYNC_TO_LDS_B128, ASYNCcnt).
// LDS byte offset = low 32 bits of the generic pointer (flat aperture, ISA 10.2).
__device__ __forceinline__ void async_ld16(void* lds_dst, const void* gsrc) {
    unsigned ldsa = (unsigned)(size_t)lds_dst;
    asm volatile("global_load_async_to_lds_b128 %0, %1, off"
                 :: "v"(ldsa), "v"(gsrc) : "memory");
}
__device__ __forceinline__ void async_wait0() {
    asm volatile("s_wait_asynccnt 0x0" ::: "memory");
}

// ---------------------------------------------------------------- prep ------
__global__ void prep_x_kernel(const float* __restrict__ tgt,
                              const float* __restrict__ qpos,
                              __bf16* __restrict__ qkin,
                              __bf16* __restrict__ xb, int n) {
    int i = blockIdx.x * blockDim.x + threadIdx.x;
    if (i < n) {
        float t = tgt[i];
        qkin[i] = (__bf16)(t + qpos[i]);
        xb[i]   = (__bf16)t;
    }
}

__global__ void prep_w_kernel(const float* __restrict__ Wq,
                              const float* __restrict__ Wk,
                              const float* __restrict__ Wv,
                              const float* __restrict__ Wo,
                              __bf16* __restrict__ outT) {
    int i = blockIdx.x * blockDim.x + threadIdx.x;
    const int total = 4 * D_ * D_;
    if (i >= total) return;
    int w = i / (D_ * D_);
    int r = i % (D_ * D_);
    int n = r / D_, k = r % D_;
    const float* W = (w == 0) ? Wq : (w == 1) ? Wk : (w == 2) ? Wv : Wo;
    outT[(size_t)w * D_ * D_ + (size_t)n * D_ + k] = (__bf16)W[(size_t)k * D_ + n];
}

// ------------------------------------------------- QKV projection GEMM ------
// grid = ((B*L)/16, 2), 256 threads = 8 waves.
//   y==0: waves 0-3 -> Q, waves 4-7 -> K, sharing the async-staged qk_in tile.
//   y==1: all 8 waves -> V (6 n-tiles each) from the staged tgt tile.
__global__ __launch_bounds__(256) void qkv_kernel(
    const __bf16* __restrict__ qkin, const __bf16* __restrict__ xb,
    const __bf16* __restrict__ wT3,   // [WqT,WkT,WvT] each [n][k] bf16
    const float* __restrict__ bq, const float* __restrict__ bk,
    const float* __restrict__ bv,
    __bf16* __restrict__ Qb,   // [B,H,L,DH]
    __bf16* __restrict__ Kb,   // [B,H,L,DH]
    __bf16* __restrict__ Vt)   // [B,H,DH,L]
{
    __shared__ __bf16 xs[16][776];                   // 16B-aligned rows, bank-rotated
    const int mode = blockIdx.y;                     // 0: Q+K, 1: V
    const int row0 = blockIdx.x * 16;                // over B*L
    const int tid = threadIdx.x, wave = tid >> 5, lane = tid & 31;
    const int lh = lane >> 4, lm = lane & 15;

    const __bf16* X = mode ? xb : qkin;
    for (int c = tid; c < 16 * (D_ / 8); c += 256) {
        const int r = c / (D_ / 8), q = c % (D_ / 8);
        async_ld16(&xs[r][q * 8], X + (size_t)(row0 + r) * D_ + q * 8);
    }
    async_wait0();
    __syncthreads();

    const int bidx = row0 / L_;
    const int l0   = row0 % L_;

    const int ntiles  = mode ? 6 : 12;
    const int isK     = mode ? 0 : (wave >> 2);
    const int colbase = mode ? wave * 96 : (wave & 3) * 192;
    const __bf16* W   = wT3 + (size_t)(mode ? 2 : isK) * D_ * D_;
    const float* bias = mode ? bv : (isK ? bk : bq);

    for (int j = 0; j < ntiles; ++j) {
        const int n0 = colbase + j * 16;
        const __bf16* wrow = W + (size_t)(n0 + lm) * D_;
        v8f acc = {};
        v16bf bcur = load_frag(wrow, lh);
        for (int kt = 0; kt < D_ / 32 - 1; ++kt) {
            v16bf bnxt = load_frag(wrow + (kt + 1) * 32, lh);   // prefetch
            v16bf a    = load_frag(&xs[lm][kt * 32], lh);       // ds path
            acc = wmma_bf16(a, bcur, acc);
            bcur = bnxt;
        }
        {
            v16bf a = load_frag(&xs[lm][(D_ / 32 - 1) * 32], lh);
            acc = wmma_bf16(a, bcur, acc);
        }
        const int col = n0 + lm;
        const int h = col >> 6, d = col & 63;
        const float bc = bias[col];
        #pragma unroll
        for (int r = 0; r < 8; ++r) {
            const int m = r + 8 * lh;
            const float val = acc[r] + bc;
            if (mode) {
                Vt[(((size_t)bidx * H_ + h) * DH_ + d) * L_ + (l0 + m)] = (__bf16)val;
            } else {
                __bf16* dst = isK ? Kb : Qb;
                dst[(((size_t)bidx * H_ + h) * L_ + (l0 + m)) * DH_ + d] = (__bf16)val;
            }
        }
    }
}

// -------------------------------------------------------- attention ---------
// grid = (L/16, H, B), 128 threads = 4 waves. Q fragments hoisted to registers,
// K fragments double-buffered over the 8 n-tiles, V double-buffered over k.
__global__ __launch_bounds__(128) void attn_kernel(
    const __bf16* __restrict__ Qb, const __bf16* __restrict__ Kb,
    const __bf16* __restrict__ Vt,
    const float* __restrict__ pl,     // [B,L,L,SPD]
    const float* __restrict__ Wloc,   // [SPD,H]
    const float* __restrict__ bloc,   // [H]
    __bf16* __restrict__ attn_out)    // [B*L, D]
{
    __shared__ float  s_tile[16][516];
    __shared__ __bf16 p_tile[16][520];               // row stride 1040B, 16B aligned

    const int m0 = blockIdx.x * 16;
    const int h  = blockIdx.y;
    const int b  = blockIdx.z;
    const int tid = threadIdx.x, wave = tid >> 5, lane = tid & 31;
    const int lh = lane >> 4, lm = lane & 15;

    const float wl0 = Wloc[0 * H_ + h], wl1 = Wloc[1 * H_ + h],
                wl2 = Wloc[2 * H_ + h], wl3 = Wloc[3 * H_ + h],
                wl4 = Wloc[4 * H_ + h];
    const float bl = bloc[h];
    const float scale = 0.125f;                      // 1/sqrt(64)

    const size_t headQK = ((size_t)b * H_ + h) * (size_t)L_ * DH_;
    const __bf16* qrow  = Qb + headQK + (size_t)(m0 + lm) * DH_;
    const __bf16* kbase = Kb + headQK;

    const v16bf aq0 = load_frag(qrow, lh);           // loop-invariant Q frags
    const v16bf aq1 = load_frag(qrow + 32, lh);

    // ---- S = scale * Q K^T + log-bias, K double-buffered ------------------
    v16bf kb0 = load_frag(kbase + (size_t)(wave * 128 + lm) * DH_, lh);
    v16bf kb1 = load_frag(kbase + (size_t)(wave * 128 + lm) * DH_ + 32, lh);
    for (int nt = 0; nt < 8; ++nt) {
        const v16bf kc0 = kb0, kc1 = kb1;
        if (nt < 7) {
            const __bf16* krow = kbase + (size_t)(wave * 128 + (nt + 1) * 16 + lm) * DH_;
            kb0 = load_frag(krow, lh);
            kb1 = load_frag(krow + 32, lh);
        }
        v8f acc = {};
        acc = wmma_bf16(aq0, kc0, acc);
        acc = wmma_bf16(aq1, kc1, acc);

        const int n = wave * 128 + nt * 16 + lm;
        #pragma unroll
        for (int r = 0; r < 8; ++r) {
            const int m = r + 8 * lh;
            const float* p = pl + (((size_t)b * L_ + (m0 + m)) * L_ + n) * SPD_;
            float dot = fmaf(p[0], wl0, fmaf(p[1], wl1,
                        fmaf(p[2], wl2, fmaf(p[3], wl3, fmaf(p[4], wl4, bl)))));
            dot = fmaxf(dot, 1e-6f);                 // relu + clip folded
            s_tile[m][n] = __logf(dot) + acc[r] * scale;
        }
    }
    __syncthreads();

    // ---- row softmax over 512 cols (8 lanes/row, butterfly reduce) --------
    const int row = tid >> 3, sub = tid & 7;
    float mx = -3.0e38f;
    for (int c = sub; c < L_; c += 8) mx = fmaxf(mx, s_tile[row][c]);
    mx = fmaxf(mx, __shfl_xor(mx, 1, 32));
    mx = fmaxf(mx, __shfl_xor(mx, 2, 32));
    mx = fmaxf(mx, __shfl_xor(mx, 4, 32));
    float sum = 0.f;
    for (int c = sub; c < L_; c += 8) {
        float e = __expf(s_tile[row][c] - mx);
        s_tile[row][c] = e;
        sum += e;
    }
    sum += __shfl_xor(sum, 1, 32);
    sum += __shfl_xor(sum, 2, 32);
    sum += __shfl_xor(sum, 4, 32);
    const float rinv = 1.0f / sum;
    for (int c = sub; c < L_; c += 8)
        p_tile[row][c] = (__bf16)(s_tile[row][c] * rinv);
    __syncthreads();

    // ---- O_head = P @ V, V double-buffered --------------------------------
    {
        const int n0 = wave * 16;
        const __bf16* vrow = Vt + ((size_t)b * H_ + h) * (size_t)DH_ * L_
                                + (size_t)(n0 + lm) * L_;
        v8f acc = {};
        v16bf vb = load_frag(vrow, lh);
        for (int kt = 0; kt < L_ / 32; ++kt) {
            const v16bf vc = vb;
            if (kt < L_ / 32 - 1) vb = load_frag(vrow + (kt + 1) * 32, lh);
            v16bf a = load_frag(&p_tile[lm][kt * 32], lh);
            acc = wmma_bf16(a, vc, acc);
        }
        const int d = n0 + lm;
        #pragma unroll
        for (int r = 0; r < 8; ++r) {
            const int m = r + 8 * lh;
            attn_out[((size_t)b * L_ + m0 + m) * D_ + h * DH_ + d] = (__bf16)acc[r];
        }
    }
}

// ------------------------------- O projection + residual + LayerNorm -------
// 256 threads = 8 waves x 6 n-tiles. LDS reused via union: bf16 A-stage
// during GEMM, fp32 row buffer for LayerNorm afterwards.
__global__ __launch_bounds__(256) void oproj_ln_kernel(
    const __bf16* __restrict__ attn_out, const __bf16* __restrict__ woT,
    const float* __restrict__ bo, const float* __restrict__ tgt,
    const float* __restrict__ ln_g, const float* __restrict__ ln_b,
    float* __restrict__ out)
{
    __shared__ union SM {
        __bf16 xs[16][776];
        float  xr[16][772];
    } sm;
    const int row0 = blockIdx.x * 16;                // over B*L
    const int tid = threadIdx.x, wave = tid >> 5, lane = tid & 31;
    const int lh = lane >> 4, lm = lane & 15;

    for (int c = tid; c < 16 * (D_ / 8); c += 256) {
        const int r = c / (D_ / 8), q = c % (D_ / 8);
        async_ld16(&sm.xs[r][q * 8], attn_out + (size_t)(row0 + r) * D_ + q * 8);
    }
    async_wait0();
    __syncthreads();

    v8f acc[6] = {};
    for (int j = 0; j < 6; ++j) {
        const int n0 = wave * 96 + j * 16;
        const __bf16* wrow = woT + (size_t)(n0 + lm) * D_;
        v16bf bcur = load_frag(wrow, lh);
        for (int kt = 0; kt < D_ / 32 - 1; ++kt) {
            v16bf bnxt = load_frag(wrow + (kt + 1) * 32, lh);
            v16bf a    = load_frag(&sm.xs[lm][kt * 32], lh);
            acc[j] = wmma_bf16(a, bcur, acc[j]);
            bcur = bnxt;
        }
        v16bf a = load_frag(&sm.xs[lm][(D_ / 32 - 1) * 32], lh);
        acc[j] = wmma_bf16(a, bcur, acc[j]);
    }
    __syncthreads();                                 // xs reads done everywhere

    #pragma unroll
    for (int j = 0; j < 6; ++j) {
        const int col = wave * 96 + j * 16 + lm;
        const float bc = bo[col];
        #pragma unroll
        for (int r = 0; r < 8; ++r) {
            const int m = r + 8 * lh;
            sm.xr[m][col] = acc[j][r] + bc + tgt[(size_t)(row0 + m) * D_ + col];
        }
    }
    __syncthreads();

    // LayerNorm: 16 lanes per row, butterfly reduce within 16-lane groups.
    const int row = tid >> 4, sub = tid & 15;
    float s = 0.f;
    for (int c = sub; c < D_; c += 16) s += sm.xr[row][c];
    s += __shfl_xor(s, 1, 32); s += __shfl_xor(s, 2, 32);
    s += __shfl_xor(s, 4, 32); s += __shfl_xor(s, 8, 32);
    const float mu = s * (1.0f / D_);
    float v = 0.f;
    for (int c = sub; c < D_; c += 16) { float d = sm.xr[row][c] - mu; v += d * d; }
    v += __shfl_xor(v, 1, 32); v += __shfl_xor(v, 2, 32);
    v += __shfl_xor(v, 4, 32); v += __shfl_xor(v, 8, 32);
    const float rstd = rsqrtf(v * (1.0f / D_) + 1e-5f);
    float* orow = out + (size_t)(row0 + row) * D_;
    for (int c = sub; c < D_; c += 16)
        orow[c] = ln_g[c] * (sm.xr[row][c] - mu) * rstd + ln_b[c];
}

// ---------------------------------------------------------------------------
extern "C" void kernel_launch(void* const* d_in, const int* in_sizes, int n_in,
                              void* d_out, int out_size, void* d_ws, size_t ws_size,
                              hipStream_t stream) {
    const float* tgt  = (const float*)d_in[0];
    const float* qpos = (const float*)d_in[1];
    const float* pl   = (const float*)d_in[2];
    const float* Wq   = (const float*)d_in[3];
    const float* bq   = (const float*)d_in[4];
    const float* Wk   = (const float*)d_in[5];
    const float* bk   = (const float*)d_in[6];
    const float* Wv   = (const float*)d_in[7];
    const float* bv   = (const float*)d_in[8];
    const float* Wo   = (const float*)d_in[9];
    const float* bo   = (const float*)d_in[10];
    const float* Wloc = (const float*)d_in[11];
    const float* bloc = (const float*)d_in[12];
    const float* ln_g = (const float*)d_in[13];
    const float* ln_b = (const float*)d_in[14];
    float* out = (float*)d_out;

    char* ws = (char*)d_ws;
    size_t off = 0;
    auto carve = [&](size_t bytes) {
        char* p = ws + off;
        off = (off + bytes + 255) & ~(size_t)255;
        return p;
    };
    const size_t NLD = (size_t)B_ * L_ * D_;
    __bf16* qkin = (__bf16*)carve(NLD * 2);                  // bf16(tgt+pos)
    __bf16* xb   = (__bf16*)carve(NLD * 2);                  // bf16(tgt)
    __bf16* wT4  = (__bf16*)carve((size_t)4 * D_ * D_ * 2);  // WqT,WkT,WvT,WoT
    __bf16* Qb   = (__bf16*)carve(NLD * 2);                  // [B,H,L,DH]
    __bf16* Kb   = (__bf16*)carve(NLD * 2);                  // [B,H,L,DH]
    __bf16* Vt   = (__bf16*)carve(NLD * 2);                  // [B,H,DH,L]
    __bf16* aout = (__bf16*)carve(NLD * 2);                  // attn output bf16
    (void)ws_size; (void)in_sizes; (void)n_in; (void)out_size;

    const int nElem = (int)NLD;
    prep_x_kernel<<<(nElem + 255) / 256, 256, 0, stream>>>(tgt, qpos, qkin, xb, nElem);
    const int wTot = 4 * D_ * D_;
    prep_w_kernel<<<(wTot + 255) / 256, 256, 0, stream>>>(Wq, Wk, Wv, Wo, wT4);

    dim3 gqkv(B_ * L_ / 16, 2);
    qkv_kernel<<<gqkv, 256, 0, stream>>>(qkin, xb, wT4, bq, bk, bv, Qb, Kb, Vt);

    dim3 gattn(L_ / 16, H_, B_);
    attn_kernel<<<gattn, 128, 0, stream>>>(Qb, Kb, Vt, pl, Wloc, bloc, aout);

    oproj_ln_kernel<<<B_ * L_ / 16, 256, 0, stream>>>(
        aout, wT4 + (size_t)3 * D_ * D_, bo, tgt, ln_g, ln_b, out);
}
